// INGP_919123001837
// MI455X (gfx1250) — matrix operand order repaired and checked by reading, not compiled, not run on hardware
//
#include <hip/hip_runtime.h>
#include <hip/hip_bf16.h>
#include <math.h>

// CDNA5 WMMA vector types
typedef _Float16 v16h __attribute__((ext_vector_type(16)));
typedef _Float16 v8h  __attribute__((ext_vector_type(8)));
typedef float    v8f  __attribute__((ext_vector_type(8)));

#define LVLS   16
#define FDIM   32            // L * F
#define HID    64
#define LOG2T  19
#define TMASK  ((1u << LOG2T) - 1u)
#define PRIME1 2654435761u
#define PRIME2 805459861u
#define WAVES  8
#define TPB    (WAVES * 32)

// floor(16 * 1.3819061^l) - 1, precomputed (f32 semantics of the reference)
__constant__ float RESM1[LVLS] = {
    15.f, 21.f, 29.f, 41.f, 57.f, 79.f, 110.f, 152.f,
    211.f, 293.f, 405.f, 560.f, 775.f, 1071.f, 1480.f, 2046.f
};

static __device__ __forceinline__ v16h cat8(v8h lo, v8h hi) {
    return __builtin_shufflevector(lo, hi, 0, 1, 2, 3, 4, 5, 6, 7,
                                           8, 9, 10, 11, 12, 13, 14, 15);
}

__global__ __launch_bounds__(TPB) void ingp_fused(
    const float* __restrict__ x,  const float* __restrict__ table,
    const float* __restrict__ w1, const float* __restrict__ b1,
    const float* __restrict__ w2, const float* __restrict__ b2,
    const float* __restrict__ w3, const float* __restrict__ b3,
    float* __restrict__ out, int n)
{
    // Transposed f16 weights ([n][k]) + biases, staged once per block
    __shared__ alignas(16) _Float16 sW1[HID][FDIM];   // 4 KB
    __shared__ alignas(16) _Float16 sW2[HID][HID];    // 8 KB
    __shared__ alignas(16) _Float16 sW3[16][HID];     // 2 KB (rows 3..15 zero)
    __shared__ float sB1[HID], sB2[HID], sB3[16];
    // Per-wave activation tiles
    __shared__ alignas(16) _Float16 sFeat[WAVES][16][FDIM]; // 8 KB
    __shared__ alignas(16) _Float16 sH1[WAVES][16][HID];    // 16 KB
    __shared__ alignas(16) _Float16 sH2[WAVES][16][HID];    // 16 KB

    const int tid = threadIdx.x;
    for (int i = tid; i < HID * FDIM; i += TPB) {
        int nn = i >> 5, kk = i & 31;
        sW1[nn][kk] = (_Float16)w1[kk * HID + nn];
    }
    for (int i = tid; i < HID * HID; i += TPB) {
        int nn = i >> 6, kk = i & 63;
        sW2[nn][kk] = (_Float16)w2[kk * HID + nn];
    }
    for (int i = tid; i < 16 * HID; i += TPB) {
        int nn = i >> 6, kk = i & 63;
        sW3[nn][kk] = (nn < 3) ? (_Float16)w3[kk * 3 + nn] : (_Float16)0.f;
    }
    if (tid < HID) { sB1[tid] = b1[tid]; sB2[tid] = b2[tid]; }
    if (tid < 16)  { sB3[tid] = (tid < 3) ? b3[tid] : 0.f; }
    __syncthreads();

    const int wv   = tid >> 5;
    const int lane = tid & 31;
    const int base = (blockIdx.x * WAVES + wv) * 16;   // first point of this wave's tile

    // ---------------- hash-grid encode: 16 pts x 16 levels, 1 (pt,lvl)/lane/iter
    #pragma unroll
    for (int it = 0; it < 8; ++it) {
        int task = it * 32 + lane;
        int pt = task >> 4, lvl = task & 15;
        int pg = base + pt; if (pg >= n) pg = n - 1;     // clamp: keeps EXEC full
        float r  = RESM1[lvl];
        float px = x[pg * 3 + 0] * r;
        float py = x[pg * 3 + 1] * r;
        float pz = x[pg * 3 + 2] * r;
        float gx = floorf(px), gy = floorf(py), gz = floorf(pz);
        float fx = px - gx, fy = py - gy, fz = pz - gz;
        unsigned ix = (unsigned)gx, iy = (unsigned)gy, iz = (unsigned)gz;
        const float* tab = table + (size_t)lvl * ((size_t)1u << LOG2T) * 2u;
        float o0 = 0.f, o1 = 0.f;
        #pragma unroll
        for (int c = 0; c < 8; ++c) {
            unsigned bi = (c >> 2) & 1u, bj = (c >> 1) & 1u, bk = c & 1u;
            unsigned h = ((ix + bi) ^ ((iy + bj) * PRIME1) ^ ((iz + bk) * PRIME2)) & TMASK;
            const float2 f2 = *(const float2*)(tab + (size_t)h * 2u);
            float wgt = (bi ? fx : 1.f - fx) * (bj ? fy : 1.f - fy) * (bk ? fz : 1.f - fz);
            o0 += wgt * f2.x;
            o1 += wgt * f2.y;
        }
        sFeat[wv][pt][2 * lvl]     = (_Float16)o0;
        sFeat[wv][pt][2 * lvl + 1] = (_Float16)o1;
    }
    // wave-private LDS region; per-wave LDS ops are in order -> no barrier needed

    // Fragment index helpers (ISA 7.12.2 layouts, wave32)
    const int  m  = lane & 15;                 // A: row, B: col, C: col
    const bool hi = (lane & 16) != 0;
    const int  ah = hi ? 8 : 0;                // A: K sub-offset per half
    const int  bh = hi ? 8 : 0;                // B: K sub-offset
    const int  mb = hi ? 8 : 0;                // C/D: row base

    // ---------------- layer 1: [16x32] @ [32x64] -> 4 WMMAs
    v16h A1;
    {
        const _Float16* rp = &sFeat[wv][m][0];
        A1 = cat8(*(const v8h*)(rp + ah), *(const v8h*)(rp + 16 + ah));
    }
    #pragma unroll
    for (int nt = 0; nt < 4; ++nt) {
        const _Float16* bp = &sW1[nt * 16 + m][0];
        v16h B = cat8(*(const v8h*)(bp + bh), *(const v8h*)(bp + 16 + bh));
        v8f c = {};
        c = __builtin_amdgcn_wmma_f32_16x16x32_f16(false, A1, false, B,
                                                   (short)0, c, false, false);
        float bias = sB1[nt * 16 + m];
        #pragma unroll
        for (int e = 0; e < 8; ++e) {
            float v = c[e] + bias;
            v = v > 0.f ? v : 0.f;
            sH1[wv][mb + e][nt * 16 + m] = (_Float16)v;
        }
    }

    // ---------------- layer 2: [16x64] @ [64x64] -> 8 WMMAs
    #pragma unroll
    for (int nt = 0; nt < 4; ++nt) {
        v8f c = {};
        #pragma unroll
        for (int kt = 0; kt < 2; ++kt) {
            int ko = kt * 32;
            const _Float16* rp = &sH1[wv][m][ko];
            v16h A = cat8(*(const v8h*)(rp + ah), *(const v8h*)(rp + 16 + ah));
            const _Float16* bp = &sW2[nt * 16 + m][ko];
            v16h B = cat8(*(const v8h*)(bp + bh), *(const v8h*)(bp + 16 + bh));
            c = __builtin_amdgcn_wmma_f32_16x16x32_f16(false, A, false, B,
                                                       (short)0, c, false, false);
        }
        float bias = sB2[nt * 16 + m];
        #pragma unroll
        for (int e = 0; e < 8; ++e) {
            float v = c[e] + bias;
            v = v > 0.f ? v : 0.f;
            sH2[wv][mb + e][nt * 16 + m] = (_Float16)v;
        }
    }

    // ---------------- layer 3: [16x64] @ [64x16(pad)] -> 2 WMMAs, sigmoid, store
    v8f c3 = {};
    #pragma unroll
    for (int kt = 0; kt < 2; ++kt) {
        int ko = kt * 32;
        const _Float16* rp = &sH2[wv][m][ko];
        v16h A = cat8(*(const v8h*)(rp + ah), *(const v8h*)(rp + 16 + ah));
        const _Float16* bp = &sW3[m][ko];
        v16h B = cat8(*(const v8h*)(bp + bh), *(const v8h*)(bp + 16 + bh));
        c3 = __builtin_amdgcn_wmma_f32_16x16x32_f16(false, A, false, B,
                                                    (short)0, c3, false, false);
    }
    {
        float bias = sB3[m];
        #pragma unroll
        for (int e = 0; e < 8; ++e) {
            if (m < 3) {
                int pt = base + mb + e;
                if (pt < n) {
                    float t = c3[e] + bias;
                    out[pt * 3 + m] = 1.f / (1.f + expf(-t));
                }
            }
        }
    }
}

extern "C" void kernel_launch(void* const* d_in, const int* in_sizes, int n_in,
                              void* d_out, int out_size, void* d_ws, size_t ws_size,
                              hipStream_t stream) {
    const float* x     = (const float*)d_in[0];
    const float* table = (const float*)d_in[1];
    const float* w1    = (const float*)d_in[2];
    const float* b1    = (const float*)d_in[3];
    const float* w2    = (const float*)d_in[4];
    const float* b2    = (const float*)d_in[5];
    const float* w3    = (const float*)d_in[6];
    const float* b3    = (const float*)d_in[7];
    float* out = (float*)d_out;

    int n      = in_sizes[0] / 3;               // 500,000 points
    int tiles  = (n + 15) / 16;                 // 16 points per wave
    int blocks = (tiles + WAVES - 1) / WAVES;   // 8 waves per block
    ingp_fused<<<blocks, TPB, 0, stream>>>(x, table, w1, b1, w2, b2, w3, b3, out, n);
}
